// DFloat11LlamaDecoderLayer_206158430636
// MI455X (gfx1250) — matrix operand order repaired and checked by reading, not compiled, not run on hardware
//
#include <hip/hip_runtime.h>
#include <hip/hip_bf16.h>

// ---------------------------------------------------------------------------
// Llama decoder layer for MI455X (gfx1250, wave32, WMMA + TDM).
// fp32 weights/activations are converted to bf16 ONCE (halves HBM traffic:
// weights are re-read 16x by GEMM blocks), then all GEMMs run
// v_wmma_f32_16x16x32_bf16 with fp32 accumulation.  Weight tiles are fetched
// by the Tensor Data Mover (tensor_load_to_lds, double-buffered, TENSORcnt
// synchronized); activation tiles use register-pipelined b128 loads.
// ---------------------------------------------------------------------------

typedef __bf16 bf16_t;
typedef bf16_t v16bf __attribute__((ext_vector_type(16)));
typedef bf16_t v8bf  __attribute__((ext_vector_type(8)));
typedef float  v8f   __attribute__((ext_vector_type(8)));
typedef unsigned int v4u __attribute__((ext_vector_type(4)));
typedef unsigned int v8u __attribute__((ext_vector_type(8)));

#define S_LEN 2048
#define H_DIM 2048
#define N_HEADS 16
#define N_KV 8
#define HEAD_D 128
#define I_DIM 8192

// flat_weights element offsets
#define OFF_QKV    0u
#define OFF_O      8388608u
#define OFF_UPGATE 12582912u
#define OFF_DOWN   46137344u
#define W_TOTAL    62914560u

// ---------------------------------------------------------------------------
// WMMA fragment helpers (layouts per CDNA5 ISA 7.12.2)
// ---------------------------------------------------------------------------
__device__ __forceinline__ v16bf ld_frag_a(const bf16_t* t, int ld, int m0,
                                           int k0, int half, int l16) {
  const bf16_t* p = t + (m0 + l16) * ld + k0 + half * 8;
  v8bf lo = *(const v8bf*)p;
  v8bf hi = *(const v8bf*)(p + 16);
  v16bf r;
#pragma unroll
  for (int i = 0; i < 8; ++i) { r[i] = lo[i]; r[i + 8] = hi[i]; }
  return r;
}

__device__ __forceinline__ v16bf ld_frag_b(const bf16_t* t, int ld, int n0,
                                           int k0, int half, int l16) {
  const bf16_t* p = t + (n0 + l16) * ld + k0 + half * 8;
  v8bf lo = *(const v8bf*)p;
  v8bf hi = *(const v8bf*)(p + 16);
  v16bf r;
#pragma unroll
  for (int i = 0; i < 8; ++i) { r[i] = lo[i]; r[i + 8] = hi[i]; }
  return r;
}

__device__ __forceinline__ v8f wmma_bf16(v16bf a, v16bf b, v8f c) {
  return __builtin_amdgcn_wmma_f32_16x16x32_bf16(
      false, a, false, b, (short)0, c, false, false);
}

// ---------------------------------------------------------------------------
// TDM: 2D tile load (128 rows x 32 bf16) global -> LDS with row padding so
// the LDS image lands with stride LDT=40 bf16 (pad 4 DWORDs per 16 DWORDs).
// D# per CDNA5 ISA 8.3/8.4.  Issued once per wave (EXEC ignored).
// ---------------------------------------------------------------------------
__device__ __forceinline__ void tdm_load_w_tile(unsigned lds_byte_off,
                                                const bf16_t* gptr,
                                                unsigned K_elems,
                                                unsigned N_rows) {
  unsigned long long ga = (unsigned long long)(uintptr_t)gptr;
  v4u g0;
  g0.x = 1u;                                      // count=1: valid user D#
  g0.y = lds_byte_off;                            // lds_addr
  g0.z = (unsigned)ga;                            // global_addr[31:0]
  g0.w = (unsigned)((ga >> 32) & 0x01FFFFFFull)   // global_addr[56:32]
         | (2u << 30);                            // type=2 ("image")
  v8u g1;
  g1[0] = (1u << 16)   // data_size = 2 bytes
        | (1u << 20)   // pad_enable
        | (3u << 22)   // pad_interval: 16 DWORDs (= one 32-elem bf16 row)
        | (3u << 25);  // pad_amount: 4 DWORDs  (-> LDS stride 40 bf16)
  g1[1] = (K_elems & 0xFFFFu) << 16;                    // tensor_dim0 lo16
  g1[2] = (K_elems >> 16) | ((N_rows & 0xFFFFu) << 16); // dim0 hi / dim1 lo
  g1[3] = (N_rows >> 16) | (32u << 16);                 // dim1 hi, tile_dim0=32
  g1[4] = 128u;                                         // tile_dim1=128
  g1[5] = K_elems;                                      // tensor_dim0_stride
  g1[6] = 0u;
  g1[7] = 0u;
  asm volatile("tensor_load_to_lds %0, %1" ::"s"(g0), "s"(g1) : "memory");
}

// ---------------------------------------------------------------------------
// fp32 -> bf16 bulk conversion (8 elems / thread, b128 in, b128 out)
// ---------------------------------------------------------------------------
__global__ __launch_bounds__(256) void cvt_f32_bf16_kernel(
    const float* __restrict__ src, bf16_t* __restrict__ dst) {
  const size_t base = ((size_t)blockIdx.x * 256 + threadIdx.x) * 8;
  float4 v0 = *(const float4*)(src + base);
  float4 v1 = *(const float4*)(src + base + 4);
  uint4 o;
  bf16_t* t = (bf16_t*)&o;
  t[0] = (bf16_t)v0.x; t[1] = (bf16_t)v0.y;
  t[2] = (bf16_t)v0.z; t[3] = (bf16_t)v0.w;
  t[4] = (bf16_t)v1.x; t[5] = (bf16_t)v1.y;
  t[6] = (bf16_t)v1.z; t[7] = (bf16_t)v1.w;
  *(uint4*)(dst + base) = o;
}

// ---------------------------------------------------------------------------
// RMSNorm: fp32 in, bf16 out. One block (256 thr) per row.
// ---------------------------------------------------------------------------
__global__ __launch_bounds__(256) void rmsnorm_kernel(
    const float* __restrict__ x, const float* __restrict__ w,
    bf16_t* __restrict__ y) {
  const int row = blockIdx.x;
  const float* xr = x + (size_t)row * H_DIM;
  float ss = 0.f;
  for (int i = threadIdx.x; i < H_DIM; i += 256) {
    float v = xr[i];
    ss = fmaf(v, v, ss);
  }
#pragma unroll
  for (int m = 16; m >= 1; m >>= 1) ss += __shfl_xor(ss, m);
  __shared__ float red[8];
  const int wave = threadIdx.x >> 5, lane = threadIdx.x & 31;
  if (lane == 0) red[wave] = ss;
  __syncthreads();
  float tot = 0.f;
#pragma unroll
  for (int i = 0; i < 8; ++i) tot += red[i];
  const float inv = rsqrtf(tot * (1.0f / H_DIM) + 1e-6f);
  bf16_t* yr = y + (size_t)row * H_DIM;
  for (int i = threadIdx.x; i < H_DIM; i += 256)
    yr[i] = (bf16_t)(w[i] * xr[i] * inv);
}

// ---------------------------------------------------------------------------
// Tiled WMMA GEMM:  C[M][N] = A[M][K] @ W[N][K]^T (+ optional fp32 residual)
// A, W: bf16.  Block 256 thr / 8 waves; tile 128x128, BK=32.
// W tile via TDM (double-buffered, issued 1 iter ahead by wave 0,
// TENSORcnt-synchronized).  A tile via register-pipelined b128 loads.
// ---------------------------------------------------------------------------
#define BM 128
#define BN 128
#define BK 32
#define LDT 40  // BK + 8 pad: 16B-aligned rows, spreads LDS banks

__global__ __launch_bounds__(256) void gemm_bf16_wmma(
    const bf16_t* __restrict__ A, const bf16_t* __restrict__ W,
    const float* __restrict__ resid, float* __restrict__ C,
    int M, int N, int K, int epi) {
  __shared__ __align__(16) bf16_t As[BM * LDT];
  __shared__ __align__(16) bf16_t Wb[2][BN * LDT];

  const int tid = threadIdx.x;
  const int lane = tid & 31;
  const int wave = tid >> 5;
  const int half = lane >> 4;
  const int l16 = lane & 15;
  const int wm = (wave & 3) * 32;
  const int wn = (wave >> 2) * 64;
  const int m0 = blockIdx.x * BM;
  const int n0 = blockIdx.y * BN;

  const int ar = tid >> 2;       // 0..63
  const int ac = (tid & 3) * 8;  // bf16 col: 0,8,16,24

  v8f acc[2][4];
#pragma unroll
  for (int i = 0; i < 2; ++i)
#pragma unroll
    for (int j = 0; j < 4; ++j) acc[i][j] = (v8f)(0.f);

  const int kt_end = K / BK;
  const bf16_t* Wtilebase = W + (size_t)n0 * K;

  // ---- prologue: TDM tile 0, A tile 0 into regs ----
  if (wave == 0)
    tdm_load_w_tile((unsigned)(uintptr_t)&Wb[0][0], Wtilebase, (unsigned)K,
                    (unsigned)N);
  uint4 pa0 = *(const uint4*)&A[(size_t)(m0 + ar) * K + ac];
  uint4 pa1 = *(const uint4*)&A[(size_t)(m0 + 64 + ar) * K + ac];

  for (int kt = 0; kt < kt_end; ++kt) {
    __syncthreads();  // prev compute done: As and Wb[(kt+1)&1] are free
    *(uint4*)&As[ar * LDT + ac] = pa0;
    *(uint4*)&As[(64 + ar) * LDT + ac] = pa1;
    if (wave == 0) {
      if (kt + 1 < kt_end) {
        tdm_load_w_tile((unsigned)(uintptr_t)&Wb[(kt + 1) & 1][0],
                        Wtilebase + (kt + 1) * BK, (unsigned)K, (unsigned)N);
        __builtin_amdgcn_s_wait_tensorcnt(1);  // tile kt complete
      } else {
        __builtin_amdgcn_s_wait_tensorcnt(0);
      }
    }
    __syncthreads();  // As stored, W[kt] landed

    if (kt + 1 < kt_end) {  // prefetch next A tile while WMMAs run
      pa0 = *(const uint4*)&A[(size_t)(m0 + ar) * K + (kt + 1) * BK + ac];
      pa1 = *(const uint4*)&A[(size_t)(m0 + 64 + ar) * K + (kt + 1) * BK + ac];
    }

    const bf16_t* Wt = Wb[kt & 1];
    v16bf af[2], bfm[4];
#pragma unroll
    for (int mt = 0; mt < 2; ++mt)
      af[mt] = ld_frag_a(As, LDT, wm + mt * 16, 0, half, l16);
#pragma unroll
    for (int nt = 0; nt < 4; ++nt)
      bfm[nt] = ld_frag_b(Wt, LDT, wn + nt * 16, 0, half, l16);
#pragma unroll
    for (int mt = 0; mt < 2; ++mt)
#pragma unroll
      for (int nt = 0; nt < 4; ++nt)
        acc[mt][nt] = wmma_bf16(af[mt], bfm[nt], acc[mt][nt]);
  }

  // ---- epilogue ----
#pragma unroll
  for (int mt = 0; mt < 2; ++mt) {
#pragma unroll
    for (int nt = 0; nt < 4; ++nt) {
      const int col = n0 + wn + nt * 16 + l16;
#pragma unroll
      for (int r = 0; r < 8; ++r) {
        const int row = m0 + wm + mt * 16 + half * 8 + r;
        float v = acc[mt][nt][r];
        if (epi) v += resid[(size_t)row * N + col];
        C[(size_t)row * N + col] = v;
      }
    }
  }
}

// ---------------------------------------------------------------------------
// RoPE over q (16 heads) and k (8 heads) in fused qkv buffer [S][4096], fp32
// ---------------------------------------------------------------------------
__global__ __launch_bounds__(256) void rope_kernel(
    float* __restrict__ qkv, const float* __restrict__ cosb,
    const float* __restrict__ sinb) {
  const int idx = blockIdx.x * 256 + threadIdx.x;  // S * 24 * 64 total
  const int d = idx & 63;
  const int head = (idx >> 6) % 24;
  const int s = idx / (24 * 64);
  const int col = (head < N_HEADS) ? head * HEAD_D
                                   : H_DIM + (head - N_HEADS) * HEAD_D;
  float* p = qkv + (size_t)s * 4096 + col;
  const float c0 = cosb[s * HEAD_D + d],      s0 = sinb[s * HEAD_D + d];
  const float c1 = cosb[s * HEAD_D + 64 + d], s1 = sinb[s * HEAD_D + 64 + d];
  const float x1 = p[d], x2 = p[d + 64];
  p[d]      = x1 * c0 - x2 * s0;
  p[d + 64] = x2 * c1 + x1 * s1;
}

// ---------------------------------------------------------------------------
// Flash attention, causal, GQA.  Block 128 thr (4 waves) = 1 head x 64 rows.
// bf16 output (feeds next GEMM directly).
// ---------------------------------------------------------------------------
#define FA_BM 64
#define FA_BN 32
#define LDQ 136
#define LDP 40

__global__ __launch_bounds__(128) void flash_attn_kernel(
    const float* __restrict__ qkv, bf16_t* __restrict__ out) {
  const int h = blockIdx.y;
  const int q0 = blockIdx.x * FA_BM;
  const int kvh = h >> 1;
  const int tid = threadIdx.x;
  const int lane = tid & 31;
  const int wave = tid >> 5;
  const int half = lane >> 4;
  const int l16 = lane & 15;

  __shared__ __align__(16) bf16_t Qs[FA_BM * LDQ];
  __shared__ __align__(16) bf16_t Ks[FA_BN * LDQ];
  __shared__ __align__(16) bf16_t Vts[HEAD_D * LDP];
  __shared__ __align__(16) bf16_t Ps[4][16 * LDP];

  for (int i = tid; i < FA_BM * 32; i += 128) {
    const int r = i >> 5;
    const int c = (i & 31) * 4;
    float4 v = *(const float4*)(&qkv[(size_t)(q0 + r) * 4096 + h * HEAD_D + c]);
    bf16_t* d = &Qs[r * LDQ + c];
    d[0] = (bf16_t)v.x; d[1] = (bf16_t)v.y;
    d[2] = (bf16_t)v.z; d[3] = (bf16_t)v.w;
  }
  __syncthreads();

  v16bf qf[4];
#pragma unroll
  for (int kc = 0; kc < 4; ++kc)
    qf[kc] = ld_frag_a(Qs, LDQ, wave * 16, kc * 32, half, l16);

  v8f o[8];
#pragma unroll
  for (int t = 0; t < 8; ++t) o[t] = (v8f)(0.f);
  float m_r[8], l_r[8];
#pragma unroll
  for (int r = 0; r < 8; ++r) { m_r[r] = -1e30f; l_r[r] = 0.f; }

  const float scale = 0.08838834764831845f;  // 1/sqrt(128)
  const int jmax = (q0 + FA_BM - 1) / FA_BN;

  for (int jb = 0; jb <= jmax; ++jb) {
    __syncthreads();
    for (int i = tid; i < FA_BN * 32; i += 128) {
      const int r = i >> 5;
      const int c = (i & 31) * 4;
      float4 v = *(const float4*)(&qkv[(size_t)(jb * FA_BN + r) * 4096 +
                                       H_DIM + kvh * HEAD_D + c]);
      bf16_t* d = &Ks[r * LDQ + c];
      d[0] = (bf16_t)v.x; d[1] = (bf16_t)v.y;
      d[2] = (bf16_t)v.z; d[3] = (bf16_t)v.w;
    }
    for (int i = tid; i < FA_BN * HEAD_D; i += 128) {
      const int r = i & 31;   // key
      const int c = i >> 5;   // dim
      float v = qkv[(size_t)(jb * FA_BN + r) * 4096 + H_DIM + N_KV * HEAD_D +
                    kvh * HEAD_D + c];
      Vts[c * LDP + r] = (bf16_t)v;
    }
    __syncthreads();

    v8f s0 = (v8f)(0.f), s1 = (v8f)(0.f);
#pragma unroll
    for (int kc = 0; kc < 4; ++kc) {
      v16bf b0 = ld_frag_b(Ks, LDQ, 0, kc * 32, half, l16);
      v16bf b1 = ld_frag_b(Ks, LDQ, 16, kc * 32, half, l16);
      s0 = wmma_bf16(qf[kc], b0, s0);
      s1 = wmma_bf16(qf[kc], b1, s1);
    }

    const int key0 = jb * FA_BN + l16;
    const int key1 = key0 + 16;
#pragma unroll
    for (int r = 0; r < 8; ++r) {
      const int rowg = q0 + wave * 16 + half * 8 + r;
      float a0 = (key0 <= rowg) ? s0[r] * scale : -1e30f;
      float a1 = (key1 <= rowg) ? s1[r] * scale : -1e30f;
      float mx = fmaxf(a0, a1);
#pragma unroll
      for (int m = 8; m >= 1; m >>= 1) mx = fmaxf(mx, __shfl_xor(mx, m));
      const float mnew = fmaxf(m_r[r], mx);
      const float p0 = __expf(a0 - mnew);
      const float p1 = __expf(a1 - mnew);
      s0[r] = p0; s1[r] = p1;
      float sum = p0 + p1;
#pragma unroll
      for (int m = 8; m >= 1; m >>= 1) sum += __shfl_xor(sum, m);
      const float alpha = __expf(m_r[r] - mnew);
      l_r[r] = l_r[r] * alpha + sum;
      m_r[r] = mnew;
#pragma unroll
      for (int t = 0; t < 8; ++t) o[t][r] *= alpha;
    }

    bf16_t* pw = &Ps[wave][0];
#pragma unroll
    for (int r = 0; r < 8; ++r) {
      pw[(half * 8 + r) * LDP + l16]      = (bf16_t)s0[r];
      pw[(half * 8 + r) * LDP + l16 + 16] = (bf16_t)s1[r];
    }
    __syncthreads();

    v16bf pf = ld_frag_a(Ps[wave], LDP, 0, 0, half, l16);
#pragma unroll
    for (int t = 0; t < 8; ++t) {
      v16bf bv = ld_frag_b(Vts, LDP, t * 16, 0, half, l16);
      o[t] = wmma_bf16(pf, bv, o[t]);
    }
  }

#pragma unroll
  for (int r = 0; r < 8; ++r) {
    const int rowg = q0 + wave * 16 + half * 8 + r;
    const float invl = 1.0f / l_r[r];
#pragma unroll
    for (int t = 0; t < 8; ++t)
      out[(size_t)rowg * H_DIM + h * HEAD_D + t * 16 + l16] =
          (bf16_t)(o[t][r] * invl);
  }
}

// ---------------------------------------------------------------------------
// h = silu(x@gate^T) * (x@up^T); gu fp32 [S][16384] = [up | gate]; bf16 out
// ---------------------------------------------------------------------------
__global__ __launch_bounds__(256) void silu_mul_kernel(
    const float* __restrict__ gu, bf16_t* __restrict__ h) {
  const size_t idx = (size_t)blockIdx.x * 256 + threadIdx.x;  // S*I total
  const size_t s = idx / I_DIM;
  const size_t i = idx - s * I_DIM;
  const float up = gu[s * (2 * I_DIM) + i];
  const float g  = gu[s * (2 * I_DIM) + I_DIM + i];
  h[idx] = (bf16_t)((g / (1.f + __expf(-g))) * up);
}

// ---------------------------------------------------------------------------
// Host-side launch sequence
// ---------------------------------------------------------------------------
extern "C" void kernel_launch(void* const* d_in, const int* in_sizes, int n_in,
                              void* d_out, int out_size, void* d_ws,
                              size_t ws_size, hipStream_t stream) {
  (void)in_sizes; (void)n_in; (void)out_size; (void)ws_size;
  const float* hidden  = (const float*)d_in[0];
  const float* flat    = (const float*)d_in[1];
  const float* in_ln   = (const float*)d_in[2];
  const float* post_ln = (const float*)d_in[3];
  const float* cosb    = (const float*)d_in[4];
  const float* sinb    = (const float*)d_in[5];
  float* out = (float*)d_out;

  // workspace layout (byte offsets, all 16B aligned)
  char* w = (char*)d_ws;
  bf16_t* wbf   = (bf16_t*)w; w += (size_t)W_TOTAL * 2;          // 125.8 MB
  float*  qkv   = (float*)w;  w += (size_t)S_LEN * 4096 * 4;     //  33.5 MB
  float*  gu    = (float*)w;  w += (size_t)S_LEN * 16384 * 4;    // 134.2 MB
  float*  hid   = (float*)w;  w += (size_t)S_LEN * H_DIM * 4;    //  16.8 MB
  bf16_t* xn    = (bf16_t*)w; w += (size_t)S_LEN * H_DIM * 2;    //   8.4 MB
  bf16_t* attnb = (bf16_t*)w; w += (size_t)S_LEN * H_DIM * 2;    //   8.4 MB
  bf16_t* hbuf  = (bf16_t*)w;                                    //  33.5 MB

  // 0) one-shot fp32 -> bf16 weight conversion
  cvt_f32_bf16_kernel<<<W_TOTAL / 2048, 256, 0, stream>>>(flat, wbf);

  // 1) input RMSNorm (bf16 out)
  rmsnorm_kernel<<<S_LEN, 256, 0, stream>>>(hidden, in_ln, xn);

  // 2) fused QKV GEMM: W rows [q(2048)|k(1024)|v(1024)] contiguous
  gemm_bf16_wmma<<<dim3(S_LEN / BM, 4096 / BN), 256, 0, stream>>>(
      xn, wbf + OFF_QKV, nullptr, qkv, S_LEN, 4096, H_DIM, 0);

  // 3) RoPE on q and k
  rope_kernel<<<(S_LEN * 24 * 64) / 256, 256, 0, stream>>>(qkv, cosb, sinb);

  // 4) causal flash attention (bf16 out)
  flash_attn_kernel<<<dim3(S_LEN / FA_BM, N_HEADS), 128, 0, stream>>>(qkv,
                                                                      attnb);

  // 5) O projection + residual
  gemm_bf16_wmma<<<dim3(S_LEN / BM, H_DIM / BN), 256, 0, stream>>>(
      attnb, wbf + OFF_O, hidden, hid, S_LEN, H_DIM, H_DIM, 1);

  // 6) post RMSNorm (bf16 out)
  rmsnorm_kernel<<<S_LEN, 256, 0, stream>>>(hid, post_ln, xn);

  // 7) fused up|gate GEMM (contiguous 16384 x 2048 weight block)
  gemm_bf16_wmma<<<dim3(S_LEN / BM, 16384 / BN), 256, 0, stream>>>(
      xn, wbf + OFF_UPGATE, nullptr, gu, S_LEN, 16384, H_DIM, 0);

  // 8) silu(gate) * up (bf16 out)
  silu_mul_kernel<<<(S_LEN * I_DIM) / 256, 256, 0, stream>>>(gu, hbuf);

  // 9) down GEMM + residual -> output
  gemm_bf16_wmma<<<dim3(S_LEN / BM, H_DIM / BN), 256, 0, stream>>>(
      hbuf, wbf + OFF_DOWN, hid, out, S_LEN, H_DIM, I_DIM, 1);
}
